// CGCNNetL1Sum_74955769249870
// MI455X (gfx1250) — matrix operand order repaired and checked by reading, not compiled, not run on hardware
//
#include <hip/hip_runtime.h>

// ---------------------------------------------------------------------------
// CGConv on MI455X (gfx1250, wave32).
// Per-edge dual GEMM (z[160] x W[160x64], twice) mapped onto
// v_wmma_f32_16x16x32_bf16: 16 edges/wave, 5 K-tiles x 4 N-tiles x 2 weights
// = 40 WMMAs per tile. bf16 inputs / f32 accumulation keeps the kernel on the
// memory roofline (~0.6 GB HBM traffic) instead of the ~0.5 ms f32-VALU path.
// ---------------------------------------------------------------------------

typedef __attribute__((ext_vector_type(16))) __bf16 v16bf;
typedef __attribute__((ext_vector_type(8)))  __bf16 v8bf;
typedef __attribute__((ext_vector_type(2)))  __bf16 v2bf;
typedef __attribute__((ext_vector_type(8)))  float  v8f;

#define F_ 64
#define D_ 32
#define KZ 160          // 2F + D
#define ZPITCH 168      // padded LDS row pitch (bf16 elems), 336B = 16B-aligned
#define NFRAG (2*4*5)   // 2 weights x 4 n-tiles x 5 k-tiles
#define WFRAG_ELEMS (NFRAG*32*16)

// --- swizzle W_f / W_s (f32, [160 x 64]) into WMMA B-fragment lane order ----
// frag(w, nt, kt): lane L, elem e  ->  K = kt*32 + (L/16)*8 + (e%8) + (e/8)*16, col = nt*16 + (L%16)
__global__ void prep_weights_kernel(const float* __restrict__ W_f,
                                    const float* __restrict__ W_s,
                                    __bf16* __restrict__ Wfrag) {
    int idx = blockIdx.x * blockDim.x + threadIdx.x;
    if (idx >= WFRAG_ELEMS) return;
    int e    = idx & 15;
    int lane = (idx >> 4) & 31;
    int kt   = (idx >> 9) % 5;
    int nt   = (idx / 2560) & 3;
    int w    = idx / 10240;
    int K    = kt * 32 + (lane >> 4) * 8 + (e & 7) + (e >> 3) * 16;
    int col  = nt * 16 + (lane & 15);
    const float* W = w ? W_s : W_f;
    Wfrag[idx] = (__bf16)W[K * F_ + col];
}

// --- zero the aggregation buffer, seed the scalar output with b_dense -------
__global__ void zero_kernel(float* __restrict__ agg, long long n,
                            float* __restrict__ out,
                            const float* __restrict__ b_dense) {
    long long i = (long long)blockIdx.x * blockDim.x + threadIdx.x;
    if (i == 0) out[0] = b_dense[0];
    long long stride = (long long)gridDim.x * blockDim.x;
    for (; i < n; i += stride) agg[i] = 0.0f;
}

// --- main edge kernel: 1 wave = 16 edges -----------------------------------
__global__ __launch_bounds__(32)
void edge_kernel(const float* __restrict__ x, const int* __restrict__ ei,
                 const float* __restrict__ ea, const __bf16* __restrict__ Wfrag,
                 const float* __restrict__ b_f, const float* __restrict__ b_s,
                 float* __restrict__ agg, int E) {
    __shared__ int s_dst[16];
    __shared__ int s_src[16];
    __shared__ __attribute__((aligned(16))) __bf16 zb[16][ZPITCH];

    const int tid = threadIdx.x;               // 0..31
    const long long e0 = (long long)blockIdx.x * 16;

    // edge_index[0][e] = src, edge_index[1][e] = dst (row-major (2,E))
    if (tid < 16) s_src[tid] = ei[e0 + tid];
    else          s_dst[tid - 16] = ei[(long long)E + e0 + (tid - 16)];
    __syncthreads();

    // Gather z rows -> LDS (bf16).  x rows: 256B coalesced (L2-resident,
    // x is 12.8MB << 192MB L2).  edge_attr: streamed 128B rows.
    for (int e = 0; e < 16; ++e) {
        const float* xd = x + (long long)s_dst[e] * F_;
        const float* xs = x + (long long)s_src[e] * F_;
        float2 vd = *(const float2*)(xd + tid * 2);
        float2 vs = *(const float2*)(xs + tid * 2);
        v2bf pd; pd.x = (__bf16)vd.x; pd.y = (__bf16)vd.y;
        v2bf ps; ps.x = (__bf16)vs.x; ps.y = (__bf16)vs.y;
        *(v2bf*)&zb[e][tid * 2]       = pd;   // cols [0,64)   = x[dst]
        *(v2bf*)&zb[e][F_ + tid * 2]  = ps;   // cols [64,128) = x[src]
        zb[e][2 * F_ + tid] = (__bf16)ea[(e0 + e) * D_ + tid]; // [128,160)
    }
    __syncthreads();

    v8f accF[4] = {};
    v8f accS[4] = {};

    // A-fragment lane mapping (16-bit A 16x32): lane = (kh<<4)|m,
    // elems 0..7 -> K = kh*8 + [0,8), elems 8..15 -> K = kh*8 + 16 + [0,8)
    const int m  = tid & 15;
    const int kh = tid >> 4;

#pragma unroll
    for (int kt = 0; kt < 5; ++kt) {
        const int kb = kt * 32 + kh * 8;
        v8bf lo = *(const v8bf*)&zb[m][kb];
        v8bf hi = *(const v8bf*)&zb[m][kb + 16];
        v16bf afrag = __builtin_shufflevector(lo, hi,
            0, 1, 2, 3, 4, 5, 6, 7, 8, 9, 10, 11, 12, 13, 14, 15);
#pragma unroll
        for (int nt = 0; nt < 4; ++nt) {
            const __bf16* pF = Wfrag + ((((0 * 4 + nt) * 5 + kt) * 32 + tid) * 16);
            const __bf16* pS = Wfrag + ((((1 * 4 + nt) * 5 + kt) * 32 + tid) * 16);
            v16bf bF = *(const v16bf*)pF;
            v16bf bS = *(const v16bf*)pS;
            accF[nt] = __builtin_amdgcn_wmma_f32_16x16x32_bf16(
                false, afrag, false, bF, (short)0, accF[nt], false, false);
            accS[nt] = __builtin_amdgcn_wmma_f32_16x16x32_bf16(
                false, afrag, false, bS, (short)0, accS[nt], false, false);
        }
    }

    // C/D layout: VGPR v, lanes 0-15 -> row v; lanes 16-31 -> row v+8; col = lane%16
    const int ncol  = tid & 15;
    const int rbase = (tid >> 4) * 8;
    int drow[8];
#pragma unroll
    for (int v = 0; v < 8; ++v) drow[v] = s_dst[rbase + v];

#pragma unroll
    for (int nt = 0; nt < 4; ++nt) {
        const int n = nt * 16 + ncol;
        const float bfv = b_f[n];
        const float bsv = b_s[n];
#pragma unroll
        for (int v = 0; v < 8; ++v) {
            float g = accF[nt][v] + bfv;
            float s = accS[nt][v] + bsv;
            float gate = 1.0f / (1.0f + __expf(-g));
            float core = (s > 20.0f) ? s : log1pf(__expf(s));
            float msg  = gate * core;
            unsafeAtomicAdd(&agg[(long long)drow[v] * F_ + n], msg);
        }
    }
}

// --- node kernel: h = relu(x+agg), sum-pool, dot with W_dense --------------
__global__ __launch_bounds__(256)
void node_kernel(const float* __restrict__ x, const float* __restrict__ agg,
                 const float* __restrict__ Wd, float* __restrict__ out, int N) {
    const int lane   = threadIdx.x & 31;
    const int warp   = blockIdx.x * (blockDim.x >> 5) + (threadIdx.x >> 5);
    const int nwarps = gridDim.x * (blockDim.x >> 5);
    float2 wd = *(const float2*)(Wd + lane * 2);
    float acc = 0.0f;
    for (long long n = warp; n < N; n += nwarps) {
        float2 xv = *(const float2*)(x   + n * F_ + lane * 2);
        float2 av = *(const float2*)(agg + n * F_ + lane * 2);
        float h0 = fmaxf(xv.x + av.x, 0.0f);
        float h1 = fmaxf(xv.y + av.y, 0.0f);
        acc += h0 * wd.x + h1 * wd.y;
    }
    // wave32 tree reduction
    for (int off = 16; off > 0; off >>= 1)
        acc += __shfl_down(acc, off, 32);
    if (lane == 0) unsafeAtomicAdd(out, acc);
}

extern "C" void kernel_launch(void* const* d_in, const int* in_sizes, int n_in,
                              void* d_out, int out_size, void* d_ws, size_t ws_size,
                              hipStream_t stream) {
    const float* x   = (const float*)d_in[0];
    const int*   ei  = (const int*)  d_in[1];
    const float* ea  = (const float*)d_in[2];
    const float* W_f = (const float*)d_in[3];
    const float* b_f = (const float*)d_in[4];
    const float* W_s = (const float*)d_in[5];
    const float* b_s = (const float*)d_in[6];
    const float* W_d = (const float*)d_in[7];
    const float* b_d = (const float*)d_in[8];
    float* out = (float*)d_out;

    const int N = in_sizes[0] / F_;
    const int E = in_sizes[1] / 2;

    // workspace layout: [0,40KB) bf16 weight fragments; [64KB, +N*F*4) agg
    __bf16* Wfrag = (__bf16*)d_ws;
    float*  agg   = (float*)((char*)d_ws + 65536);

    prep_weights_kernel<<<(WFRAG_ELEMS + 255) / 256, 256, 0, stream>>>(W_f, W_s, Wfrag);
    zero_kernel<<<512, 256, 0, stream>>>(agg, (long long)N * F_, out, b_d);
    edge_kernel<<<E / 16, 32, 0, stream>>>(x, ei, ea, Wfrag, b_f, b_s, agg, E);
    node_kernel<<<256, 256, 0, stream>>>(x, agg, W_d, out, N);
}
